// CaptioningRNN_24910810317050
// MI455X (gfx1250) — matrix operand order repaired and checked by reading, not compiled, not run on hardware
//
#include <hip/hip_runtime.h>

// Problem dims (fixed by the reference)
#define NB_   256      // batch
#define TT_   32       // caption length
#define TM1_  31       // T-1
#define VV_   10000    // vocab
#define WD_   512      // word dim
#define HH_   1024     // hidden
#define DI_   1280     // image feature dim
#define PP_   16       // spatial positions (4x4)
#define FH_   4096     // 4*H
#define KC_   2048     // concat K (h | attn)
#define ROWS_ 7936     // N*(T-1)

typedef __attribute__((ext_vector_type(16))) __bf16 v16bf;
typedef __attribute__((ext_vector_type(8)))  __bf16 v8bf;
typedef __attribute__((ext_vector_type(8)))  float  v8f;
typedef __attribute__((ext_vector_type(4)))  unsigned int v4u;
typedef __attribute__((ext_vector_type(4)))  int v4i;
typedef __attribute__((ext_vector_type(8)))  int v8i;

// ---------------------------------------------------------------------------
// WMMA fragment helpers (wave32, 16x16x32 bf16, CDNA5 ISA 7.12.2 layouts)
// ---------------------------------------------------------------------------
// A fragment: per lane, two contiguous 16-byte runs -> 2 x b128 loads.
__device__ __forceinline__ v16bf load_a_frag(const __bf16* __restrict__ A, int lda) {
  const int lane = threadIdx.x & 31;
  const int half = lane >> 4, r = lane & 15;
  const __bf16* row = A + (size_t)r * lda + half * 8;
  v8bf lo = *(const v8bf*)row;          // K = half*8 + 0..7
  v8bf hi = *(const v8bf*)(row + 16);   // K = 16 + half*8 + 0..7
  return __builtin_shufflevector(lo, hi, 0, 1, 2, 3, 4, 5, 6, 7,
                                 8, 9, 10, 11, 12, 13, 14, 15);
}

// B fragment from pre-packed layout: tile = 32 lanes x 32 contiguous bytes.
__device__ __forceinline__ v16bf load_b_frag_packed(const __bf16* __restrict__ Bp,
                                                    int tileIdx) {
  const int lane = threadIdx.x & 31;
  const v16bf* p = (const v16bf*)Bp;
  return p[(size_t)tileIdx * 32 + lane];
}

__device__ __forceinline__ v8f wmma_bf16(v16bf a, v16bf b, v8f c) {
  return __builtin_amdgcn_wmma_f32_16x16x32_bf16(false, a, false, b, (short)0, c,
                                                 false, false);
}

__device__ __forceinline__ float sigf(float x) { return 1.0f / (1.0f + expf(-x)); }

// ---------------------------------------------------------------------------
// TDM: 1-D async tensor load (contiguous, 2-byte elements) global -> LDS.
// Descriptor per CDNA5 ISA ch.8 (group0/group1; groups 2/3 zero for <=2D).
// ---------------------------------------------------------------------------
#if __has_builtin(__builtin_amdgcn_tensor_load_to_lds)
#define HAVE_TDM_ 1
__device__ __forceinline__ void tdm_load_1d_b16(void* lds_dst, const void* gsrc,
                                                unsigned int nelem) {
  const unsigned long long ga = (unsigned long long)(uintptr_t)gsrc;
  v4u g0;
  g0[0] = 1u;                                   // count=1 (valid user descriptor)
  g0[1] = (unsigned int)(uintptr_t)lds_dst;     // lds_addr (byte offset)
  g0[2] = (unsigned int)(ga & 0xFFFFFFFFu);     // global_addr[31:0]
  g0[3] = (unsigned int)((ga >> 32) & 0x1FFFFFFu) | (2u << 30);  // [56:32] | type=2
  v8i g1;
  g1[0] = (int)(1u << 16);                      // data_size=1 (2 bytes)
  g1[1] = (int)((nelem & 0xFFFFu) << 16);       // tensor_dim0[15:0]  @ bits 63:48
  g1[2] = (int)(((nelem >> 16) & 0xFFFFu) | (1u << 16)); // dim0[31:16] | tensor_dim1=1
  g1[3] = (int)((nelem & 0xFFFFu) << 16);       // tile_dim0 @ bits 127:112
  g1[4] = 1;                                    // tile_dim1 = 1
  g1[5] = (int)nelem;                           // tensor_dim0_stride[31:0]
  g1[6] = 0;                                    // stride hi / dim1_stride lo
  g1[7] = 0;
  v4i z4 = {0, 0, 0, 0};
#if __clang_major__ >= 23
  v8i z8 = {0, 0, 0, 0, 0, 0, 0, 0};
  __builtin_amdgcn_tensor_load_to_lds(g0, g1, z4, z4, z8, 0);
#else
  __builtin_amdgcn_tensor_load_to_lds(g0, g1, z4, z4, 0);
#endif
}
#else
#define HAVE_TDM_ 0
#endif

// ---------------------------------------------------------------------------
// Pack a KxN row-major f32 matrix into bf16 WMMA-B fragment order:
// out[((kt*(N/16)+nt)*32 + lane)*16 + i]  holds the v16bf each lane feeds WMMA.
// ---------------------------------------------------------------------------
__global__ void pack_b_kernel(const float* __restrict__ B, __bf16* __restrict__ Bp,
                              int K, int N) {
  const int idx = blockIdx.x * 256 + threadIdx.x;
  if (idx >= K * N) return;
  const int i    = idx & 15;
  const int lane = (idx >> 4) & 31;
  const int tile = idx >> 9;
  const int ntiles = N >> 4;
  const int kt = tile / ntiles, nt = tile % ntiles;
  const int half = lane >> 4, nn = lane & 15;
  const int kk = (i & 7) + half * 8 + (i >> 3) * 16;
  Bp[idx] = (__bf16)B[(size_t)(kt * 32 + kk) * N + nt * 16 + nn];
}

// ---------------------------------------------------------------------------
// Generic bf16 GEMM: C(MxN f32) = A(MxK bf16 row-major) * Bp(packed) [+Cin][+bias]
// One wave computes a 32x64 tile: 2 A-frags x 4 B-frags -> 8 WMMAs per K step.
// grid = (N/64, M/32), block = 32.
// ---------------------------------------------------------------------------
__global__ __launch_bounds__(32) void gemm_bf16_kernel(
    const __bf16* __restrict__ A, int lda,
    const __bf16* __restrict__ Bp, int ntiles,   // ntiles = N/16
    const float* __restrict__ Cin, int ldcin,
    const float* __restrict__ bias,
    float* __restrict__ C, int ldc, int K)
{
  const int lane = threadIdx.x;
  const int m0  = blockIdx.y * 32;
  const int nt0 = blockIdx.x * 4;              // column position in 16-wide tiles
  v8f acc[2][4] = {};
  for (int k0 = 0; k0 < K; k0 += 32) {
    const int kt = k0 >> 5;
    __builtin_prefetch(Bp + (((size_t)(kt + 1) * ntiles + nt0) << 9), 0, 3);
    v16bf a0 = load_a_frag(A + (size_t)m0 * lda + k0, lda);
    v16bf a1 = load_a_frag(A + (size_t)(m0 + 16) * lda + k0, lda);
#pragma unroll
    for (int j = 0; j < 4; ++j) {
      v16bf bf = load_b_frag_packed(Bp, kt * ntiles + nt0 + j);
      acc[0][j] = wmma_bf16(a0, bf, acc[0][j]);
      acc[1][j] = wmma_bf16(a1, bf, acc[1][j]);
    }
  }
  const int half = lane >> 4, nn = lane & 15;
#pragma unroll
  for (int j = 0; j < 4; ++j) {
    const int n = (nt0 + j) * 16 + nn;
    const float bv = bias ? bias[n] : 0.0f;
#pragma unroll
    for (int mi = 0; mi < 2; ++mi) {
#pragma unroll
      for (int v = 0; v < 8; ++v) {
        const int m = m0 + mi * 16 + v + 8 * half;
        float val = acc[mi][j][v] + bv;
        if (Cin) val += Cin[(size_t)m * ldcin + n];
        C[(size_t)m * ldc + n] = val;
      }
    }
  }
}

// ---------------------------------------------------------------------------
// Elementwise / prep kernels
// ---------------------------------------------------------------------------
// features (N, DI, 4, 4) -> featb (N*16, DI) bf16, row = n*16 + pos
__global__ void featb_kernel(const float* __restrict__ f, __bf16* __restrict__ o) {
  const int idx = blockIdx.x * 256 + threadIdx.x;  // < 4096*1280
  const int r = idx / DI_, d = idx % DI_;
  const int n = r >> 4, pos = r & 15;
  o[idx] = (__bf16)f[((size_t)n * DI_ + d) * PP_ + pos];
}

// xb[(n*31+t)*512 + k] = bf16(W_embed[captions[n][t]][k])
__global__ void embed_kernel(const int* __restrict__ cap, const float* __restrict__ We,
                             __bf16* __restrict__ xb) {
  const int idx = blockIdx.x * 256 + threadIdx.x;  // < 7936*512
  const int r = idx >> 9, k = idx & 511;
  const int n = r / TM1_, t = r % TM1_;
  const int tok = cap[n * TT_ + t];
  xb[idx] = (__bf16)We[(size_t)tok * WD_ + k];
}

// h0 = c0 = mean_p Aproj[(n*16+p)*H + k]; also seed hcat[:, 0:H] (bf16)
__global__ void h0_kernel(const float* __restrict__ Aproj, float* __restrict__ h,
                          float* __restrict__ c, __bf16* __restrict__ hcat) {
  const int idx = blockIdx.x * 256 + threadIdx.x;  // < 256*1024
  const int n = idx >> 10, k = idx & 1023;
  float s = 0.0f;
#pragma unroll
  for (int p = 0; p < PP_; ++p) s += Aproj[((size_t)n * PP_ + p) * HH_ + k];
  s *= (1.0f / 16.0f);
  h[idx] = s; c[idx] = s;
  hcat[(size_t)n * KC_ + k] = (__bf16)s;
}

// Attention: w = softmax((h . A_flat)/sqrt(H)); attn = A_flat @ w -> hcat[:,H:2H]
__global__ __launch_bounds__(256) void attn_kernel(const float* __restrict__ h,
                                                   const float* __restrict__ Aproj,
                                                   __bf16* __restrict__ hcat) {
  __shared__ float part[16][16];
  __shared__ float w[16];
  const int n = blockIdx.x, tid = threadIdx.x;
  const int p = tid & 15, g = tid >> 4;
  const float* Arow = Aproj + ((size_t)n * PP_ + p) * HH_;
  const float* hn = h + (size_t)n * HH_;
  float s = 0.0f;
  for (int k = g; k < HH_; k += 16) s += hn[k] * Arow[k];
  part[p][g] = s;
  __syncthreads();
  if (tid < 16) {
    float sc = 0.0f;
    for (int j = 0; j < 16; ++j) sc += part[tid][j];
    part[tid][0] = sc * 0.03125f;  // 1/sqrt(1024)
  }
  __syncthreads();
  if (tid < 16) {
    float m = -3.0e38f;
    for (int j = 0; j < 16; ++j) m = fmaxf(m, part[j][0]);
    part[tid][1] = expf(part[tid][0] - m);
  }
  __syncthreads();
  if (tid < 16) {
    float d = 0.0f;
    for (int j = 0; j < 16; ++j) d += part[j][1];
    w[tid] = part[tid][1] / d;
  }
  __syncthreads();
  for (int k = tid; k < HH_; k += 256) {
    float a = 0.0f;
#pragma unroll
    for (int pp = 0; pp < PP_; ++pp) a += Aproj[((size_t)n * PP_ + pp) * HH_ + k] * w[pp];
    hcat[(size_t)n * KC_ + HH_ + k] = (__bf16)a;
  }
}

// LSTM gates: pre (N,4H) -> updated c,h; also write hcat[:,0:H] and hsb row.
__global__ void gates_kernel(const float* __restrict__ pre, float* __restrict__ c,
                             float* __restrict__ h, __bf16* __restrict__ hcat,
                             __bf16* __restrict__ hsb, int t) {
  const int idx = blockIdx.x * 256 + threadIdx.x;  // < 256*1024
  const int n = idx >> 10, k = idx & 1023;
  const float* pr = pre + (size_t)n * FH_;
  const float gi = sigf(pr[k]);
  const float gf = sigf(pr[HH_ + k]);
  const float go = sigf(pr[2 * HH_ + k]);
  const float gg = tanhf(pr[3 * HH_ + k]);
  const float cn = gf * c[idx] + gi * gg;
  const float hn = go * tanhf(cn);
  c[idx] = cn; h[idx] = hn;
  hcat[(size_t)n * KC_ + k] = (__bf16)hn;
  hsb[((size_t)n * TM1_ + t) * HH_ + k] = (__bf16)hn;
}

// ---------------------------------------------------------------------------
// Fused vocab GEMM + online log-softmax + NLL.  grid = 496 x 256 threads.
// 16 rows staged in LDS (TDM async tensor load when available); 8 waves sweep
// 125 chunks of 5 column tiles (625 = 5*125).
// ---------------------------------------------------------------------------
__global__ __launch_bounds__(256) void vocab_loss_kernel(
    const __bf16* __restrict__ hsb, const __bf16* __restrict__ Wvp,
    const float* __restrict__ bvocab, const int* __restrict__ cap,
    float* __restrict__ blockSums)
{
  __shared__ alignas(32) __bf16 sA[16 * HH_];   // 32 KB row tile
  __shared__ float sD[8][16][16];               // per-wave D staging
  __shared__ float sWM[8][16], sWS[8][16], sWT[8][16];
  __shared__ float sNll[16];
  const int tid = threadIdx.x;
  const int wid = tid >> 5, lane = tid & 31;
  const int row0 = blockIdx.x * 16;

#if HAVE_TDM_
  // Async tensor DMA of the contiguous 16x1024 bf16 tile, issued by wave 0.
  if (wid == 0) {
    tdm_load_1d_b16(sA, hsb + (size_t)row0 * HH_, 16u * HH_);
    __builtin_amdgcn_s_wait_tensorcnt(0);
  }
  __syncthreads();
#else
  {
    const uint4* src = (const uint4*)(hsb + (size_t)row0 * HH_);
    uint4* dst = (uint4*)sA;
    for (int i = tid; i < (16 * HH_) / 8; i += 256) dst[i] = src[i];
  }
  __syncthreads();
#endif

  const int half = lane >> 4, nn = lane & 15;
  const int gr = row0 + (lane & 15);
  const int nS = gr / TM1_, tS = gr % TM1_;
  const int tok = cap[nS * TT_ + tS + 1];

  float runM = -3.0e38f, runS = 0.0f, runT = 0.0f;

  for (int chunk = wid; chunk < 125; chunk += 8) {
    const int tbase = chunk * 5;
    v8f acc[5] = {};
    for (int k0 = 0; k0 < HH_; k0 += 32) {
      const int kt = k0 >> 5;
      __builtin_prefetch(Wvp + (((size_t)(kt + 1) * 625 + tbase) << 9), 0, 3);
      v16bf af = load_a_frag(sA + k0, HH_);   // ds_load_b128 pair
#pragma unroll
      for (int j = 0; j < 5; ++j) {
        v16bf bf = load_b_frag_packed(Wvp, kt * 625 + tbase + j);
        acc[j] = wmma_bf16(af, bf, acc[j]);
      }
    }
#pragma unroll
    for (int j = 0; j < 5; ++j) {
      const int col0 = (tbase + j) * 16;
      const float bv = bvocab[col0 + nn];
#pragma unroll
      for (int v = 0; v < 8; ++v) sD[wid][v + 8 * half][nn] = acc[j][v] + bv;
      asm volatile("s_wait_dscnt 0" ::: "memory");   // in-wave LDS RAW fence
      if (lane < 16) {
        float x[16];
        float tm = -3.0e38f;
#pragma unroll
        for (int q = 0; q < 16; ++q) { x[q] = sD[wid][lane][q]; tm = fmaxf(tm, x[q]); }
        const float newM = fmaxf(runM, tm);
        float s = runS * expf(runM - newM);
#pragma unroll
        for (int q = 0; q < 16; ++q) {
          s += expf(x[q] - newM);
          if (col0 + q == tok) runT += x[q];
        }
        runM = newM; runS = s;
      }
    }
  }
  if (lane < 16) { sWM[wid][lane] = runM; sWS[wid][lane] = runS; sWT[wid][lane] = runT; }
  __syncthreads();
  if (tid < 16) {
    float m = -3.0e38f;
    for (int wv = 0; wv < 8; ++wv) m = fmaxf(m, sWM[wv][tid]);
    float s = 0.0f, tg = 0.0f;
    for (int wv = 0; wv < 8; ++wv) { s += sWS[wv][tid] * expf(sWM[wv][tid] - m); tg += sWT[wv][tid]; }
    const float lse = m + logf(s);
    const int g2 = row0 + tid;
    const int n2 = g2 / TM1_, t2 = g2 % TM1_;
    const int tk = cap[n2 * TT_ + t2 + 1];
    sNll[tid] = (tk != 0) ? (lse - tg) : 0.0f;
  }
  __syncthreads();
  if (tid == 0) {
    float s = 0.0f;
    for (int r = 0; r < 16; ++r) s += sNll[r];
    blockSums[blockIdx.x] = s;
  }
}

__global__ void final_reduce_kernel(const float* __restrict__ bs, float* __restrict__ out) {
  float s = 0.0f;
  for (int i = 0; i < ROWS_ / 16; ++i) s += bs[i];
  out[0] = s * (1.0f / (float)NB_);
}

// ---------------------------------------------------------------------------
extern "C" void kernel_launch(void* const* d_in, const int* in_sizes, int n_in,
                              void* d_out, int out_size, void* d_ws, size_t ws_size,
                              hipStream_t stream) {
  (void)in_sizes; (void)n_in; (void)out_size; (void)ws_size;
  const float* features = (const float*)d_in[0];
  const int*   captions = (const int*)  d_in[1];
  const float* W_proj   = (const float*)d_in[2];
  const float* b_proj   = (const float*)d_in[3];
  const float* W_embed  = (const float*)d_in[4];
  const float* Wx       = (const float*)d_in[5];
  const float* Wh       = (const float*)d_in[6];
  const float* Wattn    = (const float*)d_in[7];
  const float* b        = (const float*)d_in[8];
  const float* W_vocab  = (const float*)d_in[9];
  const float* b_vocab  = (const float*)d_in[10];
  float* out = (float*)d_out;

  char* ws = (char*)d_ws;
  size_t off = 0;
  auto alloc = [&](size_t bytes) -> void* {
    void* p = ws + off;
    off = (off + bytes + 255) & ~(size_t)255;
    return p;
  };
  __bf16* featb  = (__bf16*)alloc((size_t)4096 * DI_ * 2);
  __bf16* Wpp    = (__bf16*)alloc((size_t)DI_ * HH_ * 2);    // packed
  float*  Aproj  = (float*) alloc((size_t)4096 * HH_ * 4);
  __bf16* Wxp    = (__bf16*)alloc((size_t)WD_ * FH_ * 2);    // packed
  __bf16* Whcatp = (__bf16*)alloc((size_t)KC_ * FH_ * 2);    // packed [Wh; Wattn]
  __bf16* Wvp    = (__bf16*)alloc((size_t)HH_ * VV_ * 2);    // packed
  __bf16* xb     = (__bf16*)alloc((size_t)ROWS_ * WD_ * 2);
  float*  xWx    = (float*) alloc((size_t)ROWS_ * FH_ * 4);
  float*  hbuf   = (float*) alloc((size_t)NB_ * HH_ * 4);
  float*  cbuf   = (float*) alloc((size_t)NB_ * HH_ * 4);
  __bf16* hcat   = (__bf16*)alloc((size_t)NB_ * KC_ * 2);
  float*  pre    = (float*) alloc((size_t)NB_ * FH_ * 4);
  __bf16* hsb    = (__bf16*)alloc((size_t)ROWS_ * HH_ * 2);
  float*  blkSum = (float*) alloc((size_t)(ROWS_ / 16) * 4);

  // --- one-time weight conversion + fragment packing (fp32 -> packed bf16) ---
  pack_b_kernel<<<(DI_ * HH_ + 255) / 256, 256, 0, stream>>>(W_proj, Wpp, DI_, HH_);
  pack_b_kernel<<<(WD_ * FH_ + 255) / 256, 256, 0, stream>>>(Wx, Wxp, WD_, FH_);
  pack_b_kernel<<<(HH_ * FH_ + 255) / 256, 256, 0, stream>>>(Wh, Whcatp, HH_, FH_);
  pack_b_kernel<<<(HH_ * FH_ + 255) / 256, 256, 0, stream>>>(
      Wattn, Whcatp + (size_t)(HH_ / 32) * (FH_ / 16) * 512, HH_, FH_);
  pack_b_kernel<<<(HH_ * VV_ + 255) / 256, 256, 0, stream>>>(W_vocab, Wvp, HH_, VV_);
  featb_kernel<<<(4096 * DI_) / 256, 256, 0, stream>>>(features, featb);
  embed_kernel<<<(ROWS_ * WD_) / 256, 256, 0, stream>>>(captions, W_embed, xb);

  // --- projection: Aproj (4096 x 1024) = featb (4096 x 1280) * W_proj + b_proj ---
  gemm_bf16_kernel<<<dim3(HH_ / 64, 4096 / 32), 32, 0, stream>>>(
      featb, DI_, Wpp, HH_ / 16, nullptr, 0, b_proj, Aproj, HH_, DI_);

  // --- hoisted input GEMM: xWx (7936 x 4096) = xb * Wx ---
  gemm_bf16_kernel<<<dim3(FH_ / 64, ROWS_ / 32), 32, 0, stream>>>(
      xb, WD_, Wxp, FH_ / 16, nullptr, 0, nullptr, xWx, FH_, WD_);

  // --- initial state ---
  h0_kernel<<<(NB_ * HH_) / 256, 256, 0, stream>>>(Aproj, hbuf, cbuf, hcat);

  // --- LSTM scan over 31 steps ---
  for (int t = 0; t < TM1_; ++t) {
    attn_kernel<<<NB_, 256, 0, stream>>>(hbuf, Aproj, hcat);
    gemm_bf16_kernel<<<dim3(FH_ / 64, NB_ / 32), 32, 0, stream>>>(
        hcat, KC_, Whcatp, FH_ / 16, xWx + (size_t)t * FH_, TM1_ * FH_, b, pre, FH_, KC_);
    gates_kernel<<<(NB_ * HH_) / 256, 256, 0, stream>>>(pre, cbuf, hbuf, hcat, hsb, t);
  }

  // --- fused vocab GEMM + log-softmax + NLL, deterministic reduction ---
  vocab_loss_kernel<<<ROWS_ / 16, 256, 0, stream>>>(hsb, Wvp, b_vocab, captions, blkSum);
  final_reduce_kernel<<<1, 1, 0, stream>>>(blkSum, out);
}
